// HypergraphTransformer_54391465836691
// MI455X (gfx1250) — matrix operand order repaired and checked by reading, not compiled, not run on hardware
//
#include <hip/hip_runtime.h>
#include <math.h>

// ---------------------------------------------------------------------------
// CDNA5 (gfx1250) WMMA types
// ---------------------------------------------------------------------------
typedef __attribute__((ext_vector_type(16))) __bf16 v16bf;
typedef __attribute__((ext_vector_type(8)))  __bf16 v8bf;
typedef __attribute__((ext_vector_type(4)))  __bf16 v4bf;
typedef __attribute__((ext_vector_type(8)))  float  v8f;

#define BM 128
#define BN 64
#define BK 64

// Build a 16-element bf16 fragment from two contiguous 16B LDS chunks.
__device__ __forceinline__ v16bf load_frag2(const __bf16* p0, const __bf16* p1) {
  v8bf lo = *(const v8bf*)p0;
  v8bf hi = *(const v8bf*)p1;
  v16bf r;
#pragma unroll
  for (int i = 0; i < 8; ++i) { r[i] = lo[i]; r[i + 8] = hi[i]; }
  return r;
}

// A fragment (16x32, ISA 7.12.2): lane L -> row = rt + (L&15); lanes<16 carry
// K = kk+[0..7] and kk+[16..23], lanes>=16 carry K offset +8. As layout: [row][k],
// row stride BK, contiguous K => two ds_load_b128.
__device__ __forceinline__ v16bf a_frag(const __bf16* As, int rt, int kk, int lane) {
  const __bf16* p = As + (rt + (lane & 15)) * BK + kk + ((lane >> 4) << 3);
  return load_frag2(p, p + 16);
}

// B fragment (32x16): lane L -> col = ct + (L&15); lanes<16 carry K=kk+[0..7]
// and kk+[16..23], lanes>=16 K offset +8. Bs stored transposed: [col][k], stride BK.
__device__ __forceinline__ v16bf b_frag(const __bf16* Bs, int ct, int kk, int lane) {
  const __bf16* p = Bs + (ct + (lane & 15)) * BK + kk + ((lane >> 4) << 3);
  return load_frag2(p, p + 16);
}

__device__ __forceinline__ float epilogue_apply(float x, int r, int c, int ldc,
                                                const float* Cadd, const float* bias,
                                                int mode, float scale) {
  switch (mode) {
    case 1:  return x + bias[c];                                 // + bias row-vec
    case 2:  return x + Cadd[(size_t)r * ldc + c];               // + addend matrix
    case 3:  return __expf(x) * scale;                           // exp * scale (phi)
    case 4: { float y = x + bias[c]; return y > 0.f ? y : 0.f; } // bias + relu
    default: return x;
  }
}

#define WMMA_BF16(A_, B_, C_) \
  __builtin_amdgcn_wmma_f32_16x16x32_bf16(false, (A_), false, (B_), (short)0, (C_), false, false)

// C = op(A) @ B (+epilogue). f32 in/out, bf16 WMMA with f32 accumulation.
// Requires: M % 128 == 0, N % 64 == 0, K % 64 == 0 (true at all call sites).
__global__ void __launch_bounds__(256)
wmma_gemm_kernel(const float* __restrict__ A, const float* __restrict__ B,
                 const float* __restrict__ Cadd, const float* __restrict__ bias,
                 float* __restrict__ Cout,
                 int M, int N, int Kd, int lda, int ldb, int ldc,
                 int transA, int epi, float scale)
{
  __shared__ alignas(16) __bf16 As[BM * BK];   // [row][k]
  __shared__ alignas(16) __bf16 Bs[BN * BK];   // [col][k]  (transposed tile)
  const int m0 = blockIdx.y * BM, n0 = blockIdx.x * BN;
  const int tid  = threadIdx.x;
  const int wave = tid >> 5, lane = tid & 31;
  const int rt = (wave >> 1) * 32;   // wave's 32-row strip inside block tile
  const int ct = (wave & 1) * 32;    // wave's 32-col strip

  v8f acc00 = {}; v8f acc01 = {}; v8f acc10 = {}; v8f acc11 = {};

  for (int k0 = 0; k0 < Kd; k0 += BK) {
    // ---- stage A tile (f32 -> bf16) ----
    if (!transA) {
      // row-major: global_load_b128 along K, ds_store_b64 of packed bf16x4
#pragma unroll
      for (int i = 0; i < (BM * BK) / (256 * 4); ++i) {   // 8 float4 per thread
        int idx = tid + i * 256;
        int r  = idx >> 4;              // 16 float4 per row
        int c4 = (idx & 15) << 2;
        float4 f = *(const float4*)(A + (size_t)(m0 + r) * lda + k0 + c4);
        v4bf o;
        o[0] = (__bf16)f.x; o[1] = (__bf16)f.y; o[2] = (__bf16)f.z; o[3] = (__bf16)f.w;
        *(v4bf*)(As + r * BK + c4) = o;  // ds_store_b64
      }
    } else {
      // A^T: thread owns (row r, 4 consecutive K). Each of the 4 loads is
      // coalesced across lanes (r is the lane-fast dim); pack -> one ds_store_b64.
#pragma unroll
      for (int i = 0; i < (BM * BK) / (256 * 4); ++i) {   // 8 groups per thread
        int idx = tid + i * 256;
        int r = idx & (BM - 1);         // lane-fast: coalesced global reads
        int k = (idx >> 7) << 2;
        const float* ap = A + (size_t)(k0 + k) * lda + m0 + r;
        float x0 = ap[0];
        float x1 = ap[(size_t)lda];
        float x2 = ap[2 * (size_t)lda];
        float x3 = ap[3 * (size_t)lda];
        v4bf o;
        o[0] = (__bf16)x0; o[1] = (__bf16)x1; o[2] = (__bf16)x2; o[3] = (__bf16)x3;
        *(v4bf*)(As + r * BK + k) = o;
      }
    }
    // ---- stage B tile into [col][k]: thread owns (col c, 4 consecutive K) ----
#pragma unroll
    for (int i = 0; i < (BN * BK) / (256 * 4); ++i) {     // 4 groups per thread
      int idx = tid + i * 256;
      int c = idx & (BN - 1);           // lane-fast: coalesced global reads
      int k = (idx >> 6) << 2;
      const float* bp = B + (size_t)(k0 + k) * ldb + n0 + c;
      float x0 = bp[0];
      float x1 = bp[(size_t)ldb];
      float x2 = bp[2 * (size_t)ldb];
      float x3 = bp[3 * (size_t)ldb];
      v4bf o;
      o[0] = (__bf16)x0; o[1] = (__bf16)x1; o[2] = (__bf16)x2; o[3] = (__bf16)x3;
      *(v4bf*)(Bs + c * BK + k) = o;    // ds_store_b64
    }
    if (k0 + BK < Kd) {   // CDNA5 global_prefetch_b8 of next tiles
      __builtin_prefetch(&B[(size_t)(k0 + BK + (tid >> 6)) * ldb + n0 + (tid & 63)], 0, 1);
      if (!transA) __builtin_prefetch(&A[(size_t)(m0 + (tid >> 1)) * lda + k0 + BK + ((tid & 1) << 5)], 0, 1);
      else         __builtin_prefetch(&A[(size_t)(k0 + BK + (tid >> 6)) * lda + m0 + ((tid & 63) << 1)], 0, 1);
    }
    __syncthreads();

#pragma unroll
    for (int kk = 0; kk < BK; kk += 32) {
      v16bf a0 = a_frag(As, rt,      kk, lane);
      v16bf a1 = a_frag(As, rt + 16, kk, lane);
      v16bf b0 = b_frag(Bs, ct,      kk, lane);
      v16bf b1 = b_frag(Bs, ct + 16, kk, lane);
      acc00 = WMMA_BF16(a0, b0, acc00);
      acc01 = WMMA_BF16(a0, b1, acc01);
      acc10 = WMMA_BF16(a1, b0, acc10);
      acc11 = WMMA_BF16(a1, b1, acc11);
    }
    __syncthreads();
  }

  // C/D layout: VGPR v -> row M = v + 8*(lane>=16); col N = lane&15
  const int rb0 = m0 + rt + ((lane >> 4) << 3);
  const int cb0 = n0 + ct + (lane & 15);
#pragma unroll
  for (int v = 0; v < 8; ++v) {
    int r0 = rb0 + v, r1 = rb0 + 16 + v;
    Cout[(size_t)r0 * ldc + cb0]      = epilogue_apply(acc00[v], r0, cb0,      ldc, Cadd, bias, epi, scale);
    Cout[(size_t)r0 * ldc + cb0 + 16] = epilogue_apply(acc01[v], r0, cb0 + 16, ldc, Cadd, bias, epi, scale);
    Cout[(size_t)r1 * ldc + cb0]      = epilogue_apply(acc10[v], r1, cb0,      ldc, Cadd, bias, epi, scale);
    Cout[(size_t)r1 * ldc + cb0 + 16] = epilogue_apply(acc11[v], r1, cb0 + 16, ldc, Cadd, bias, epi, scale);
  }
}

// ---------------------------------------------------------------------------
// Elementwise / reduction kernels (one 128-thread block per row, D = 128)
// ---------------------------------------------------------------------------
__global__ void gather_embed_kernel(const int* __restrict__ ids,
                                    const float* __restrict__ embed,
                                    float* __restrict__ out) {
  int i = blockIdx.x, d = threadIdx.x;
  out[(size_t)i * 128 + d] = embed[(size_t)ids[i] * 128 + d];
}

__global__ void rownorm_kernel(const float* __restrict__ X, float* __restrict__ Y) {
  __shared__ float sb[128];
  int r = blockIdx.x, d = threadIdx.x;
  float v = X[(size_t)r * 128 + d];
  sb[d] = v * v; __syncthreads();
  for (int s = 64; s > 0; s >>= 1) { if (d < s) sb[d] += sb[d + s]; __syncthreads(); }
  float inv = 1.f / fmaxf(sqrtf(sb[0]), 1e-6f);
  Y[(size_t)r * 128 + d] = v * inv;
}

__global__ void row_sumnorm_kernel(float* __restrict__ X) {
  __shared__ float sb[128];
  int r = blockIdx.x, d = threadIdx.x;
  float v = X[(size_t)r * 128 + d];
  sb[d] = v; __syncthreads();
  for (int s = 64; s > 0; s >>= 1) { if (d < s) sb[d] += sb[d + s]; __syncthreads(); }
  X[(size_t)r * 128 + d] = v / fmaxf(sb[0], 1e-6f);
}

__global__ void add_ln_kernel(float* __restrict__ Z, const float* __restrict__ Xa,
                              const float* __restrict__ g, const float* __restrict__ b) {
  __shared__ float sb[128];
  int r = blockIdx.x, d = threadIdx.x;
  float h = Z[(size_t)r * 128 + d] + Xa[(size_t)r * 128 + d];
  sb[d] = h; __syncthreads();
  for (int s = 64; s > 0; s >>= 1) { if (d < s) sb[d] += sb[d + s]; __syncthreads(); }
  float mu = sb[0] * (1.f / 128.f);
  __syncthreads();
  float c = h - mu;
  sb[d] = c * c; __syncthreads();
  for (int s = 64; s > 0; s >>= 1) { if (d < s) sb[d] += sb[d + s]; __syncthreads(); }
  float var = sb[0] * (1.f / 128.f);
  Z[(size_t)r * 128 + d] = c * rsqrtf(var + 1e-5f) * g[d] + b[d];
}

// One wave32 per edge; 4 dims/lane; shfl_xor tree; float atomic scatter into ATT.
__global__ void __launch_bounds__(256)
edge_kernel(const int* __restrict__ src, const int* __restrict__ dst,
            const float* __restrict__ Q, const float* __restrict__ K,
            const float* __restrict__ pQ, const float* __restrict__ pK,
            const float* __restrict__ V, float* __restrict__ ATT,
            int E, float inv_sd) {
  int e = blockIdx.x * 8 + (threadIdx.x >> 5);
  if (e >= E) return;
  int lane = threadIdx.x & 31;
  int is = src[e], id = dst[e];
  const float* q  = Q  + (size_t)is * 128;
  const float* k  = K  + (size_t)id * 128;
  const float* fq = pQ + (size_t)is * 128;
  const float* fk = pK + (size_t)id * 128;
  const float* v  = V  + (size_t)id * 128;
  float s = 0.f;
#pragma unroll
  for (int j = 0; j < 4; ++j) {
    int d = lane + 32 * j;
    s += __expf(q[d] * k[d] * inv_sd) - fq[d] * fk[d];
  }
#pragma unroll
  for (int off = 16; off > 0; off >>= 1) s += __shfl_xor(s, off, 32);
  float* out = ATT + (size_t)is * 128;
#pragma unroll
  for (int j = 0; j < 4; ++j) {
    int d = lane + 32 * j;
    atomicAdd(&out[d], s * v[d]);
  }
}

__global__ void pool_head_kernel(const float* __restrict__ Z,
                                 const float* __restrict__ W,
                                 const float* __restrict__ b,
                                 float* __restrict__ out, int n) {
  __shared__ float sb[128];
  int d = threadIdx.x;
  float acc = 0.f;
  for (int i = 0; i < n; ++i) acc += Z[(size_t)i * 128 + d];
  sb[d] = (acc / (float)n) * W[d];
  __syncthreads();
  for (int s = 64; s > 0; s >>= 1) { if (d < s) sb[d] += sb[d + s]; __syncthreads(); }
  if (d == 0) out[0] = sb[0] + b[0];
}

// ---------------------------------------------------------------------------
// Host side
// ---------------------------------------------------------------------------
static void run_gemm(hipStream_t st, const float* A, const float* B,
                     const float* Cadd, const float* bias, float* Cout,
                     int M, int N, int K, int lda, int ldb, int ldc,
                     int transA, int epi, float scale) {
  dim3 grid(N / BN, M / BM);
  wmma_gemm_kernel<<<grid, 256, 0, st>>>(A, B, Cadd, bias, Cout,
                                         M, N, K, lda, ldb, ldc, transA, epi, scale);
}

// params flattened in setup_inputs() insertion order:
// 0 Wq, 1 bq, 2 Wk, 3 bk, 4 Wv, 5 bv, 6 Wo, 7 bo,
// 8 ffn_W1, 9 ffn_b1, 10 ffn_W2, 11 ffn_b2, 12 ln1_g, 13 ln1_b, 14 ln2_g, 15 ln2_b
static void ksa_layer(hipStream_t st, void* const* d_in, int pbase, float* Z,
                      const float* omega, const int* idx_src, const int* idx_dst,
                      int NM, int E,
                      float* Q, float* Kb, float* V, float* Qn, float* Kn,
                      float* phiQ, float* phiK, float* ATT, float* FFNH, float* S) {
  const int D = 128, R = 128;
  const float inv_sr = 0.08838834764831845f; // 1/sqrt(128)
  const float inv_sd = 0.08838834764831845f;
  #define PP(i) ((const float*)d_in[pbase + (i)])
  run_gemm(st, Z, PP(0), nullptr, PP(1), Q,  NM, D, D, D, D, D, 0, 1, 0.f);
  run_gemm(st, Z, PP(2), nullptr, PP(3), Kb, NM, D, D, D, D, D, 0, 1, 0.f);
  run_gemm(st, Z, PP(4), nullptr, PP(5), V,  NM, D, D, D, D, D, 0, 1, 0.f);
  rownorm_kernel<<<NM, 128, 0, st>>>(Q, Qn);
  rownorm_kernel<<<NM, 128, 0, st>>>(Kb, Kn);
  run_gemm(st, Qn, omega, nullptr, nullptr, phiQ, NM, R, D, D, R, R, 0, 3, inv_sr);
  run_gemm(st, Kn, omega, nullptr, nullptr, phiK, NM, R, D, D, R, R, 0, 3, inv_sr);
  // S = phiK^T @ V   [R, D]
  run_gemm(st, phiK, V, nullptr, nullptr, S, R, D, NM, R, D, D, 1, 0, 0.f);
  // ATT = phiQ @ S   (low-rank term)
  run_gemm(st, phiQ, S, nullptr, nullptr, ATT, NM, D, R, R, D, D, 0, 0, 0.f);
  edge_kernel<<<(E + 7) / 8, 256, 0, st>>>(idx_src, idx_dst, Q, Kb, phiQ, phiK, V, ATT, E, inv_sd);
  row_sumnorm_kernel<<<NM, 128, 0, st>>>(ATT);
  run_gemm(st, ATT, PP(6), nullptr, PP(7), Qn /*attn_out*/, NM, D, D, D, D, D, 0, 1, 0.f);
  add_ln_kernel<<<NM, 128, 0, st>>>(Z, Qn, PP(12), PP(13));
  run_gemm(st, Z, PP(8), nullptr, PP(9), FFNH, NM, 4 * D, D, D, 4 * D, 4 * D, 0, 4, 0.f);
  run_gemm(st, FFNH, PP(10), nullptr, PP(11), Kn /*ffn_out*/, NM, D, 4 * D, 4 * D, D, D, 0, 1, 0.f);
  add_ln_kernel<<<NM, 128, 0, st>>>(Z, Kn, PP(14), PP(15));
  #undef PP
}

extern "C" void kernel_launch(void* const* d_in, const int* in_sizes, int n_in,
                              void* d_out, int out_size, void* d_ws, size_t ws_size,
                              hipStream_t stream) {
  const int D = 128;
  const int*   token_ids = (const int*)  d_in[0];
  const float* H         = (const float*)d_in[1];
  const float* X_E       = (const float*)d_in[2];
  const float* embed_W   = (const float*)d_in[3];
  const float* W_V       = (const float*)d_in[4];
  const float* W_E       = (const float*)d_in[5];
  const float* omega     = (const float*)d_in[6];
  // params1 -> d_in[7..22], params2 -> d_in[23..38]
  const float* mlp_W     = (const float*)d_in[39];
  const float* mlp_b     = (const float*)d_in[40];
  const int*   idx_src   = (const int*)  d_in[41];
  const int*   idx_dst   = (const int*)  d_in[42];

  const int n  = in_sizes[0];
  const int m  = in_sizes[2] / D;
  const int NM = n + m;
  const int E  = in_sizes[41];

  // workspace layout (f32)
  float* w = (float*)d_ws;
  size_t off = 0;
  auto alloc = [&](size_t elems) { float* p = w + off; off += elems; return p; };
  const size_t NMD = (size_t)NM * D;
  float* Z    = alloc(NMD);
  float* XV   = alloc((size_t)n * D);
  float* Q    = alloc(NMD);
  float* Kb   = alloc(NMD);
  float* V    = alloc(NMD);
  float* Qn   = alloc(NMD);
  float* Kn   = alloc(NMD);
  float* phiQ = alloc(NMD);
  float* phiK = alloc(NMD);
  float* ATT  = alloc(NMD);
  float* FFNH = alloc((size_t)NM * 4 * D);
  float* S    = alloc((size_t)D * D);

  // Z = [ embed[token_ids] + H @ W_V ; X_E + H^T @ W_E ]
  gather_embed_kernel<<<n, 128, 0, stream>>>(token_ids, embed_W, XV);
  run_gemm(stream, H, W_V, XV,  nullptr, Z,                 n, D, m, m, D, D, 0, 2, 0.f);
  run_gemm(stream, H, W_E, X_E, nullptr, Z + (size_t)n * D, m, D, n, m, D, D, 1, 2, 0.f);

  ksa_layer(stream, d_in, 7,  Z, omega, idx_src, idx_dst, NM, E,
            Q, Kb, V, Qn, Kn, phiQ, phiK, ATT, FFNH, S);
  ksa_layer(stream, d_in, 23, Z, omega, idx_src, idx_dst, NM, E,
            Q, Kb, V, Qn, Kn, phiQ, phiK, ATT, FFNH, S);

  pool_head_kernel<<<1, 128, 0, stream>>>(Z, mlp_W, mlp_b, (float*)d_out, n);
}
